// DCNLayer_83167746720269
// MI455X (gfx1250) — matrix operand order repaired and checked by reading, not compiled, not run on hardware
//
#include <hip/hip_runtime.h>
#include <math.h>

// ---------------------------------------------------------------------------
// DCNv3 block for MI455X (gfx1250, wave32, WMMA + Tensor Data Mover).
// GEMMs in bf16 via v_wmma_f32_16x16x32_bf16, A-tiles DMA'd to LDS with
// tensor_load_to_lds (TENSORcnt double-buffered); LN/dwconv/DCN in f32 VALU.
// ---------------------------------------------------------------------------

typedef __bf16 bf16;
typedef __attribute__((ext_vector_type(16))) __bf16 v16bf;
typedef __attribute__((ext_vector_type(8)))  __bf16 v8bf;
typedef __attribute__((ext_vector_type(8)))  float  v8f;
typedef __attribute__((ext_vector_type(4)))  unsigned int u32x4;
typedef __attribute__((ext_vector_type(8)))  int i32x8;
typedef __attribute__((ext_vector_type(4)))  int i32x4;

static constexpr int Bn = 8, Hn = 56, Wn = 56, Cn = 256, Gn = 16, Pn = 9;
static constexpr int NTOK = Bn * Hn * Wn;          // 25088 = 98 * 256
static constexpr int CM   = 4 * Cn;                // 1024
static constexpr int NOM  = 448;                   // 288 offsets + 144 logits, padded
static constexpr int LDSA = 40;                    // LDS pitch in halves (80B = 20 dwords)

#if __has_builtin(__builtin_amdgcn_tensor_load_to_lds)
#define USE_TDM 1
#else
#define USE_TDM 0
#endif

__device__ __forceinline__ float geluf(float x) {
    return 0.5f * x * (1.0f + erff(x * 0.70710678118654752440f));
}
__device__ __forceinline__ bf16  f2bf(float f) { return (bf16)f; }
__device__ __forceinline__ float bf2f(bf16 h)  { return (float)h; }

// -------- block-wide sum over 256 threads (8 wave32) -----------------------
__device__ __forceinline__ float blockSum256(float v, float* sh) {
    #pragma unroll
    for (int o = 16; o > 0; o >>= 1) v += __shfl_xor(v, o, 32);
    int w = threadIdx.x >> 5;
    if ((threadIdx.x & 31) == 0) sh[w] = v;
    __syncthreads();
    if (threadIdx.x < 8) {
        float t = sh[threadIdx.x];
        #pragma unroll
        for (int o = 4; o > 0; o >>= 1) t += __shfl_xor(t, o, 8);
        if (threadIdx.x == 0) sh[0] = t;
    }
    __syncthreads();
    float r = sh[0];
    __syncthreads();
    return r;
}

// ---------------------------------------------------------------------------
// Weight pre-pack: row-major f32 [K x (N1|N2|pad)] -> bf16 WMMA B-tile layout.
//   lane<16 : N = nt*16+lane,    K = kt*32 + i        (i = 0..15)
//   lane>=16: N = nt*16+lane-16, K = kt*32 + 16 + i
// ---------------------------------------------------------------------------
__global__ void pack_b_kernel(const float* __restrict__ s1, int N1,
                              const float* __restrict__ s2, int N2,
                              bf16* __restrict__ dst, int K, int Ndst) {
    int ntiles = Ndst >> 4;
    size_t total = (size_t)K * Ndst;
    size_t idx = (size_t)blockIdx.x * 256 + threadIdx.x;
    if (idx >= total) return;
    int i    = (int)(idx & 15);
    int lane = (int)((idx >> 4) & 31);
    size_t rest = idx >> 9;
    int nt = (int)(rest % ntiles);
    int kt = (int)(rest / ntiles);
    int n = nt * 16 + (lane & 15);
    int k = kt * 32 + ((lane >> 4) << 4) + i;
    float v = 0.0f;
    if (n < N1)            v = s1[(size_t)k * N1 + n];
    else if (n < N1 + N2)  v = s2[(size_t)k * N2 + (n - N1)];
    dst[idx] = f2bf(v);
}

__global__ void pack_bias_om_kernel(const float* __restrict__ boff,
                                    const float* __restrict__ bm,
                                    float* __restrict__ dst) {
    int i = blockIdx.x * 256 + threadIdx.x;
    if (i >= NOM) return;
    float v = 0.0f;
    if (i < 288)      v = boff[i];
    else if (i < 432) v = bm[i - 288];
    dst[i] = v;
}

// ---------------------------------------------------------------------------
// LayerNorm over C=256, one token per 256-thread block.
// ---------------------------------------------------------------------------
__global__ void ln_kernel(const float* __restrict__ x,
                          const float* __restrict__ g, const float* __restrict__ b,
                          float* __restrict__ outF, bf16* __restrict__ outB) {
    __shared__ float red[8];
    size_t base = (size_t)blockIdx.x * Cn;
    int c = threadIdx.x;
    float v = x[base + c];
    float mean = blockSum256(v, red) * (1.0f / Cn);
    float d = v - mean;
    float var = blockSum256(d * d, red) * (1.0f / Cn);
    float o = d * rsqrtf(var + 1e-6f) * g[c] + b[c];
    if (outF) outF[base + c] = o;
    if (outB) outB[base + c] = f2bf(o);
}

// ---------------------------------------------------------------------------
// dw 3x3 (C=256) + bias + LN + exact GELU -> bf16. One token/block.
// ---------------------------------------------------------------------------
__global__ void dwconv_ln_gelu_kernel(const float* __restrict__ h,
                                      const float* __restrict__ k9,
                                      const float* __restrict__ kb,
                                      const float* __restrict__ g, const float* __restrict__ b,
                                      bf16* __restrict__ out) {
    __shared__ float red[8];
    int tok = blockIdx.x;
    int bb = tok / (Hn * Wn), rem = tok % (Hn * Wn);
    int y = rem / Wn, x = rem % Wn;
    int c = threadIdx.x;
    float acc = kb[c];
    #pragma unroll
    for (int ky = 0; ky < 3; ++ky) {
        int yy = y + ky - 1;
        if (yy < 0 || yy >= Hn) continue;
        #pragma unroll
        for (int kx = 0; kx < 3; ++kx) {
            int xx = x + kx - 1;
            if (xx < 0 || xx >= Wn) continue;
            acc += h[(((size_t)bb * Hn + yy) * Wn + xx) * Cn + c] * k9[(ky * 3 + kx) * Cn + c];
        }
    }
    float mean = blockSum256(acc, red) * (1.0f / Cn);
    float d = acc - mean;
    float var = blockSum256(d * d, red) * (1.0f / Cn);
    float o = geluf(d * rsqrtf(var + 1e-6f) * g[c] + b[c]);
    out[(size_t)tok * Cn + c] = f2bf(o);
}

// ---------------------------------------------------------------------------
// dw 3x3 (C=1024) + bias + GELU on bf16. One token/block, 4 ch/thread.
// ---------------------------------------------------------------------------
__global__ void dwconv_gelu_kernel(const bf16* __restrict__ yIn,
                                   const float* __restrict__ k9,
                                   const float* __restrict__ kb,
                                   bf16* __restrict__ out) {
    int tok = blockIdx.x;
    int bb = tok / (Hn * Wn), rem = tok % (Hn * Wn);
    int y = rem / Wn, x = rem % Wn;
    #pragma unroll
    for (int i = 0; i < 4; ++i) {
        int c = i * 256 + threadIdx.x;
        float acc = kb[c];
        #pragma unroll
        for (int ky = 0; ky < 3; ++ky) {
            int yy = y + ky - 1;
            if (yy < 0 || yy >= Hn) continue;
            #pragma unroll
            for (int kx = 0; kx < 3; ++kx) {
                int xx = x + kx - 1;
                if (xx < 0 || xx >= Wn) continue;
                acc += bf2f(yIn[(((size_t)bb * Hn + yy) * Wn + xx) * CM + c]) * k9[(ky * 3 + kx) * CM + c];
            }
        }
        out[(size_t)tok * CM + c] = f2bf(geluf(acc));
    }
}

// ---------------------------------------------------------------------------
// DCNv3 core: one token per block; LDS-staged coords + per-group softmax.
// ---------------------------------------------------------------------------
__global__ void dcn_kernel(const float* __restrict__ xp,
                           const float* __restrict__ om,
                           bf16* __restrict__ out) {
    __shared__ float sOff[288];
    __shared__ float sLog[144];
    __shared__ int   sX0[144], sY0[144];
    __shared__ float sFx[144], sFy[144], sMw[144];
    int tok = blockIdx.x;
    int bb = tok / (Hn * Wn), rem = tok % (Hn * Wn);
    int ty = rem / Wn, tx = rem % Wn;
    int t = threadIdx.x;

    const float* omr = om + (size_t)tok * NOM;
    for (int i = t; i < 432; i += 256) {
        if (i < 288) sOff[i] = omr[i];
        else         sLog[i - 288] = omr[i];
    }
    __syncthreads();

    if (t < 144) {
        int p = t % 9;
        int g = t / 9;
        float ox = sOff[t * 2 + 0], oy = sOff[t * 2 + 1];
        float gx = (float)(p / 3) - 1.0f;     // x outer
        float gy = (float)(p % 3) - 1.0f;     // y inner
        float px = (float)tx + 1.0f + gx + ox;   // padded-map pixel coords
        float py = (float)ty + 1.0f + gy + oy;
        float x0 = floorf(px), y0 = floorf(py);
        sX0[t] = (int)x0; sY0[t] = (int)y0;
        sFx[t] = px - x0; sFy[t] = py - y0;
        const float* lg = sLog + g * 9;
        float mx = lg[0];
        #pragma unroll
        for (int i = 1; i < 9; ++i) mx = fmaxf(mx, lg[i]);
        float sum = 0.0f;
        #pragma unroll
        for (int i = 0; i < 9; ++i) sum += expf(lg[i] - mx);
        sMw[t] = expf(lg[p] - mx) / sum;
    }
    __syncthreads();

    int g = t >> 4, c = t & 15;
    const float* base = xp + (size_t)bb * Hn * Wn * Cn + g * 16 + c;
    float acc = 0.0f;
    #pragma unroll
    for (int p = 0; p < 9; ++p) {
        int gp = g * 9 + p;
        int x0 = sX0[gp], y0 = sY0[gp];
        float fx = sFx[gp], fy = sFy[gp];
        float s = 0.0f;
        #pragma unroll
        for (int oy = 0; oy < 2; ++oy) {
            #pragma unroll
            for (int ox = 0; ox < 2; ++ox) {
                int ry = y0 + oy - 1, rx = x0 + ox - 1;   // padded -> real
                if (ry >= 0 && ry < Hn && rx >= 0 && rx < Wn) {
                    float w = (oy ? fy : 1.0f - fy) * (ox ? fx : 1.0f - fx);
                    s += w * base[((size_t)ry * Wn + rx) * Cn];
                }
            }
        }
        acc += sMw[gp] * s;
    }
    out[(size_t)tok * Cn + g * 16 + c] = f2bf(acc);
}

// ---------------------------------------------------------------------------
// WMMA bf16 GEMM: out = epilogue(A[M,K] @ B[K,N] + bias).
// Block = 256 thr = 8 waves; block tile 256x64; wave tile 32x64 (2x4 accum,
// 8 wmma per 32-K step). A tile (256x32 bf16) is DMA'd into LDS by the
// Tensor Data Mover with hardware row padding (pitch 20 dwords = LDSA halves),
// double-buffered on TENSORcnt so tile k+1 streams under tile k's WMMAs.
// Requires M % 256 == 0, N % 64 == 0, K % 32 == 0 (all satisfied here).
// ---------------------------------------------------------------------------
__global__ __launch_bounds__(256)
void wmma_gemm_kernel(const bf16* __restrict__ A,
                      const bf16* __restrict__ Bp,
                      const float* __restrict__ bias,
                      const float* __restrict__ res,
                      float* __restrict__ outF,
                      bf16* __restrict__ outB,
                      int M, int N, int K, int actGelu) {
    __shared__ bf16 As[2][256 * LDSA];
    int t = threadIdx.x;
    int lane = t & 31, wave = t >> 5;
    int rowBlk = blockIdx.x * 256;
    int ntiles = N >> 4;
    int nt0 = blockIdx.y * 4;
    int ktiles = K >> 5;

    v8f acc[2][4] = {};

    int lrow  = wave * 32 + (lane & 15);
    int lhalf = (lane >> 4) << 4;            // 0 or 16

#if USE_TDM
    // Issue one TDM DMA for the 256-row x 64B A tile of k-step `kt` into
    // LDS buffer `buf`. D# per CDNA5 ISA ch.8: data_size=4B, tile 16x256
    // dwords, tensor stride K/2 dwords, pad 4 dwords every 16 (-> 80B pitch).
    auto tdm_issue = [&](int kt, int buf) {
        unsigned lds = (unsigned)(uintptr_t)(&As[buf][0]);   // low 32 = LDS offset
        unsigned long long ga =
            (unsigned long long)(uintptr_t)(A + (size_t)rowBlk * K + (size_t)kt * 32);
        u32x4 g0;
        g0[0] = 1u;                                          // count=1 (valid)
        g0[1] = lds;                                         // lds_addr
        g0[2] = (unsigned)ga;                                // global_addr[31:0]
        g0[3] = (unsigned)((ga >> 32) & 0x1ffffffu) | (2u << 30);  // addr[56:32]|type=2
        i32x8 g1;
        g1[0] = (2 << 16) | (1 << 20) | (3 << 22) | (3 << 25); // dsz=4B,pad_en,int=16dw,amt=4dw
        g1[1] = (16 << 16);                                  // tensor_dim0 = 16 dwords
        g1[2] = (256 << 16);                                 // tensor_dim1 = 256 rows
        g1[3] = (16 << 16);                                  // tile_dim0 = 16 dwords
        g1[4] = 256;                                         // tile_dim1 = 256 rows
        g1[5] = (K >> 1);                                    // dim0 stride (dwords)
        g1[6] = 0;
        g1[7] = 0;
        i32x4 gz = {0, 0, 0, 0};
#if __clang_major__ >= 23
        i32x8 gz8 = {0, 0, 0, 0, 0, 0, 0, 0};
        __builtin_amdgcn_tensor_load_to_lds(g0, g1, gz, gz, gz8, 0);
#else
        __builtin_amdgcn_tensor_load_to_lds(g0, g1, gz, gz, 0);
#endif
    };
    if (t == 0) tdm_issue(0, 0);
#else
    auto stage = [&](int kt, int buf) {
        #pragma unroll
        for (int i = 0; i < 4; ++i) {
            int idx = t + i * 256;           // 0..1023
            int r   = idx >> 2;              // row 0..255
            int ch  = idx & 3;               // 16B chunk
            const bf16* src = A + (size_t)(rowBlk + r) * K + kt * 32 + ch * 8;
            *(v8bf*)&As[buf][r * LDSA + ch * 8] = *(const v8bf*)src;
        }
    };
    (void)stage;
#endif

    for (int kt = 0; kt < ktiles; ++kt) {
        int buf = kt & 1;
#if USE_TDM
        if (kt + 1 < ktiles) {
            if (t == 0) tdm_issue(kt + 1, buf ^ 1);
            __builtin_amdgcn_s_wait_tensorcnt(1);   // tile kt landed; kt+1 in flight
        } else {
            __builtin_amdgcn_s_wait_tensorcnt(0);
        }
        asm volatile("" ::: "memory");              // LDS written by DMA, not by us
        __syncthreads();
#else
        stage(kt, buf);
        __syncthreads();
#endif
        const bf16* asb = &As[buf][0];
        // A fragments in contiguous K order (lanes<16: K[0..15], lanes>=16: K[16..31])
        v8bf a0lo = *(const v8bf*)&asb[(size_t)lrow * LDSA + lhalf];
        v8bf a0hi = *(const v8bf*)&asb[(size_t)lrow * LDSA + 8 + lhalf];
        v8bf a1lo = *(const v8bf*)&asb[(size_t)(lrow + 16) * LDSA + lhalf];
        v8bf a1hi = *(const v8bf*)&asb[(size_t)(lrow + 16) * LDSA + 8 + lhalf];
        v16bf af0 = __builtin_shufflevector(a0lo, a0hi,
                      0, 1, 2, 3, 4, 5, 6, 7, 8, 9, 10, 11, 12, 13, 14, 15);
        v16bf af1 = __builtin_shufflevector(a1lo, a1hi,
                      0, 1, 2, 3, 4, 5, 6, 7, 8, 9, 10, 11, 12, 13, 14, 15);

        #pragma unroll
        for (int nt = 0; nt < 4; ++nt) {
            v16bf bfrag = *(const v16bf*)(Bp +
                ((size_t)((size_t)kt * ntiles + nt0 + nt) * 32 + lane) * 16);
            acc[0][nt] = __builtin_amdgcn_wmma_f32_16x16x32_bf16(
                false, af0, false, bfrag, (short)0, acc[0][nt], false, false);
            acc[1][nt] = __builtin_amdgcn_wmma_f32_16x16x32_bf16(
                false, af1, false, bfrag, (short)0, acc[1][nt], false, false);
        }
        __syncthreads();
    }

    // Epilogue: D layout — VGPR j: lanes0-15 M=j, lanes16-31 M=8+j.
    int colLane = lane & 15;
    int rsel = (lane >> 4) * 8;
    #pragma unroll
    for (int a = 0; a < 2; ++a) {
        #pragma unroll
        for (int nt = 0; nt < 4; ++nt) {
            int col = (nt0 + nt) * 16 + colLane;
            float bv = bias ? bias[col] : 0.0f;
            #pragma unroll
            for (int j = 0; j < 8; ++j) {
                int row = rowBlk + wave * 32 + a * 16 + rsel + j;
                float v = acc[a][nt][j] + bv;
                if (actGelu) v = geluf(v);
                size_t o = (size_t)row * N + col;
                if (res)  v += res[o];
                if (outF) outF[o] = v;
                if (outB) outB[o] = f2bf(v);
            }
        }
    }
}

// ---------------------------------------------------------------------------
extern "C" void kernel_launch(void* const* d_in, const int* in_sizes, int n_in,
                              void* d_out, int out_size, void* d_ws, size_t ws_size,
                              hipStream_t stream) {
    const float* x    = (const float*)d_in[0];
    const float* g1   = (const float*)d_in[1];
    const float* b1   = (const float*)d_in[2];
    const float* Wi   = (const float*)d_in[3];
    const float* bi   = (const float*)d_in[4];
    const float* dwk  = (const float*)d_in[5];
    const float* dwb  = (const float*)d_in[6];
    const float* gdw  = (const float*)d_in[7];
    const float* bdw  = (const float*)d_in[8];
    const float* Woff = (const float*)d_in[9];
    const float* boff = (const float*)d_in[10];
    const float* Wm   = (const float*)d_in[11];
    const float* bm   = (const float*)d_in[12];
    const float* Wo   = (const float*)d_in[13];
    const float* bo   = (const float*)d_in[14];
    const float* g2   = (const float*)d_in[15];
    const float* b2   = (const float*)d_in[16];
    const float* w1   = (const float*)d_in[17];
    const float* dwk2 = (const float*)d_in[18];
    const float* dwb2 = (const float*)d_in[19];
    const float* w2   = (const float*)d_in[20];
    float* out = (float*)d_out;

    char* ws = (char*)d_ws;
    size_t off = 0;
    auto take = [&](size_t bytes) -> char* {
        char* p = ws + off;
        off += (bytes + 255) & ~(size_t)255;
        return p;
    };
    float* hF   = (float*)take((size_t)NTOK * Cn * 4);
    bf16*  hB   = (bf16*) take((size_t)NTOK * Cn * 2);
    float* xpF  = (float*)take((size_t)NTOK * Cn * 4);
    bf16*  x1B  = (bf16*) take((size_t)NTOK * Cn * 2);
    float* omF  = (float*)take((size_t)NTOK * NOM * 4);
    bf16*  dcnB = (bf16*) take((size_t)NTOK * Cn * 2);
    bf16*  h2B  = (bf16*) take((size_t)NTOK * Cn * 2);
    bf16*  y1B  = (bf16*) take((size_t)NTOK * CM * 2);
    bf16*  y2B  = (bf16*) take((size_t)NTOK * CM * 2);
    bf16*  wiP  = (bf16*) take((size_t)Cn * Cn * 2);
    bf16*  womP = (bf16*) take((size_t)Cn * NOM * 2);
    bf16*  woP  = (bf16*) take((size_t)Cn * Cn * 2);
    bf16*  w1P  = (bf16*) take((size_t)Cn * CM * 2);
    bf16*  w2P  = (bf16*) take((size_t)CM * Cn * 2);
    float* bomF = (float*)take((size_t)NOM * 4);
    (void)ws_size; (void)in_sizes; (void)n_in; (void)out_size;

    dim3 blk(256);
    auto packGrid = [](size_t total) { return dim3((unsigned)((total + 255) / 256)); };

    // ---- weight packing (bf16, WMMA tile layout) ----
    pack_b_kernel<<<packGrid((size_t)Cn * Cn), blk, 0, stream>>>(Wi, Cn, nullptr, 0, wiP, Cn, Cn);
    pack_b_kernel<<<packGrid((size_t)Cn * NOM), blk, 0, stream>>>(Woff, 288, Wm, 144, womP, Cn, NOM);
    pack_b_kernel<<<packGrid((size_t)Cn * Cn), blk, 0, stream>>>(Wo, Cn, nullptr, 0, woP, Cn, Cn);
    pack_b_kernel<<<packGrid((size_t)Cn * CM), blk, 0, stream>>>(w1, CM, nullptr, 0, w1P, Cn, CM);
    pack_b_kernel<<<packGrid((size_t)CM * Cn), blk, 0, stream>>>(w2, Cn, nullptr, 0, w2P, CM, Cn);
    pack_bias_om_kernel<<<dim3(2), blk, 0, stream>>>(boff, bm, bomF);

    const int MB = NTOK / 256;   // 98

    // ---- sub-block 1: x + DCNv3(LN(x)) ----
    ln_kernel<<<dim3(NTOK), blk, 0, stream>>>(x, g1, b1, hF, hB);
    wmma_gemm_kernel<<<dim3(MB, Cn / 64), blk, 0, stream>>>(   // xp = h@Wi + bi
        hB, wiP, bi, nullptr, xpF, nullptr, NTOK, Cn, Cn, 0);
    dwconv_ln_gelu_kernel<<<dim3(NTOK), blk, 0, stream>>>(hF, dwk, dwb, gdw, bdw, x1B);
    wmma_gemm_kernel<<<dim3(MB, NOM / 64), blk, 0, stream>>>(  // [offset | mask logits]
        x1B, womP, bomF, nullptr, omF, nullptr, NTOK, NOM, Cn, 0);
    dcn_kernel<<<dim3(NTOK), blk, 0, stream>>>(xpF, omF, dcnB);
    wmma_gemm_kernel<<<dim3(MB, Cn / 64), blk, 0, stream>>>(   // d_out = x + dcn@Wo + bo
        dcnB, woP, bo, x, out, nullptr, NTOK, Cn, Cn, 0);

    // ---- sub-block 2: x + FFN(LN(x)) ----
    ln_kernel<<<dim3(NTOK), blk, 0, stream>>>(out, g2, b2, nullptr, h2B);
    wmma_gemm_kernel<<<dim3(MB, CM / 64), blk, 0, stream>>>(   // y = GELU(h2@w1)
        h2B, w1P, nullptr, nullptr, nullptr, y1B, NTOK, CM, Cn, 1);
    dwconv_gelu_kernel<<<dim3(NTOK), blk, 0, stream>>>(y1B, dwk2, dwb2, y2B);
    wmma_gemm_kernel<<<dim3(MB, Cn / 64), blk, 0, stream>>>(   // out += y2@w2
        y2B, w2P, nullptr, out, out, nullptr, NTOK, Cn, CM, 0);
}
